// SNN_2d_lsnn_front_84078279786798
// MI455X (gfx1250) — matrix operand order repaired and checked by reading, not compiled, not run on hardware
//
#include <hip/hip_runtime.h>

// ---- CDNA5 / gfx1250 fused SNN front-end: conv3x3 + BN (folded) + 5-step LIF scan
// One pass over memory: read input once, write spikes once (~336 MB @ 23.3 TB/s ~ 14 us).
// Conv done as implicit GEMM on v_wmma_f32_16x16x32_f16:
//   M = 16 output pixels (a row segment), N = 16 (channels; 2 real), K = 32 (C*3*3=18, padded).
// Each wave owns one (b, h, w0..w0+15) segment for ALL 5 timesteps, so the LIF
// recurrence stays in registers in the WMMA C/D layout (lane = channel, comp = pixel).

typedef __attribute__((ext_vector_type(16))) _Float16 v16h;
typedef __attribute__((ext_vector_type(8)))  float    v8f;

#define DECAYF  0.2f
#define THRESHF 0.3f
#define BETAF   0.07f
#define BN_EPSF 1e-5f

#define Bn 16
#define Sn 5
#define Cn 2
#define On 2
#define Hn 512
#define Wn 512
#define SEG_W 16
#define SEGS_PER_ROW (Wn / SEG_W)          // 32
#define WAVES_PER_BLOCK 8
#define PATCH (Sn * Cn * 3 * 18)           // 540 f16 halo elements per wave
#define WAVE_LDS 576                       // halves per wave region (incl. zero slot @540)

__global__ __launch_bounds__(256)
void snn_front_kernel(const float* __restrict__ in,    // [B,S,C,H,W]
                      const float* __restrict__ cw,    // [O,C,3,3]
                      const float* __restrict__ cb,    // [O]
                      const float* __restrict__ gmm,   // [O]
                      const float* __restrict__ btb,   // [O]
                      const float* __restrict__ rmn,   // [O]
                      const float* __restrict__ rvr,   // [O]
                      const float* __restrict__ rho,   // [O]
                      float* __restrict__ out)         // [B,10,H,W] flat (s*2+o raw reshape)
{
    __shared__ _Float16 lds[WAVES_PER_BLOCK * WAVE_LDS];

    const int tid  = threadIdx.x;
    const int lane = tid & 31;
    const int wave = tid >> 5;
    const int hi   = lane >> 4;            // half-wave select (WMMA row/K split)
    const int n    = lane & 15;            // WMMA column: channel (real for n<2)
    _Float16* wl = lds + wave * WAVE_LDS;

    // ---- segment = one (b, h, 16-wide pixel row) owned by this wave
    const int seg = blockIdx.x * WAVES_PER_BLOCK + wave;   // grid sized exactly: no bounds check
    const int b   = seg / (Hn * SEGS_PER_ROW);
    const int r2  = seg % (Hn * SEGS_PER_ROW);
    const int h   = r2 / SEGS_PER_ROW;
    const int w0  = (r2 % SEGS_PER_ROW) * SEG_W;

    // ---- fold BN into conv weights/bias; per-lane channel state
    float biasf = 0.0f, adec = 0.0f;
    _Float16 wfh[18];
    if (n < On) {
        const float invs = gmm[n] * rsqrtf(rvr[n] + BN_EPSF);
        biasf = (cb[n] - rmn[n]) * invs + btb[n];
        const float rc = fminf(fmaxf(rho[n], 0.032f), 0.055f);
        adec = expf(-0.05f / rc);
#pragma unroll
        for (int k = 0; k < 18; ++k) wfh[k] = (_Float16)(cw[n * 18 + k] * invs);
    } else {
#pragma unroll
        for (int k = 0; k < 18; ++k) wfh[k] = (_Float16)0.0f;
    }

    // ---- B operand (32x16, K x N): lane = column n (+16 for K-half), halves = K run.
    // lanes 0-15 carry K=0..15, lanes 16-31 carry K=16..31 (K>=18 padded with 0).
    v16h Bm;
#pragma unroll
    for (int i = 0; i < 16; ++i) {
        _Float16 v = hi ? ((i == 0) ? wfh[16] : (i == 1) ? wfh[17] : (_Float16)0.0f)
                        : wfh[i];
        Bm[i] = v;
    }

    // ---- A operand gather map: 16-bit A 16x32, contiguous K order (same as B).
    // lane<16: m=lane, halves i -> K=i
    // lane>=16: m=lane-16, halves i -> K=16+i   (K>=18 -> zero slot)
    int offv[16];
#pragma unroll
    for (int i = 0; i < 16; ++i) {
        const int k = i + hi * 16;
        if (k < 18) {
            const int c  = k / 9;
            const int rr = (k % 9) / 3;
            const int dc = k % 3;
            offv[i] = (c * 3 + rr) * 18 + (n + dc);   // m == lane&15 == n
        } else {
            offv[i] = -1;
        }
    }

    // ---- stage f32 halo patch -> f16 LDS: [s][c][3 rows][18 cols], zero-padded borders
    const float* inb = in + (long)b * (Sn * Cn) * (Hn * (long)Wn);
    for (int e = lane; e < PATCH; e += 32) {
        const int col = e % 18;
        const int t   = e / 18;       // (s*2+c)*3 + rr
        const int rr  = t % 3;
        const int sc  = t / 3;        // s*2+c
        const int hh  = h - 1 + rr;
        const int ww  = w0 - 1 + col;
        float v = 0.0f;
        if ((unsigned)hh < (unsigned)Hn && (unsigned)ww < (unsigned)Wn)
            v = inb[(long)sc * (Hn * (long)Wn) + (long)hh * Wn + ww];
        wl[e] = (_Float16)v;
    }
    if (lane == 0) wl[PATCH] = (_Float16)0.0f;      // zero slot for K-padding reads
    __builtin_amdgcn_wave_barrier();                 // per-wave LDS: DS pipe is in-order

    // ---- LIF state in WMMA C/D layout: comp i <-> pixel m = i + 8*hi, lane -> channel n
    v8f mem = {0.f,0.f,0.f,0.f,0.f,0.f,0.f,0.f};
    v8f aad = {0.f,0.f,0.f,0.f,0.f,0.f,0.f,0.f};
    const long hw    = (long)Hn * Wn;
    const long obase = (long)b * 10 * hw + (long)h * Wn + w0 + hi * 8;

#pragma unroll
    for (int s = 0; s < Sn; ++s) {
        // gather A row for this timestep from LDS (invalid K -> zero slot)
        const int sbase = s * (Cn * 3 * 18);
        v16h A;
#pragma unroll
        for (int i = 0; i < 16; ++i) {
            const int idx = (offv[i] < 0) ? PATCH : (sbase + offv[i]);
            A[i] = wl[idx];
        }

        // conv + folded BN: D = A x B + bias   (EXEC is all-ones here: uniform flow)
        v8f acc;
#pragma unroll
        for (int i = 0; i < 8; ++i) acc[i] = biasf;
        acc = __builtin_amdgcn_wmma_f32_16x16x32_f16(
            /*neg_a=*/false, A, /*neg_b=*/false, Bm,
            /*c_mod=*/(short)0, acc, /*reuse_a=*/false, /*reuse_b=*/false);

        // LIF update (elementwise on accumulator layout)
        v8f spk;
#pragma unroll
        for (int i = 0; i < 8; ++i) {
            const float Ath = THRESHF + BETAF * aad[i];
            const float mv  = mem[i] + acc[i];
            const float sp  = (mv > Ath) ? 1.0f : 0.0f;
            mem[i] = mv * DECAYF * (1.0f - sp);
            aad[i] = adec * aad[i] + sp;
            spk[i] = sp;
        }

        // store spikes: out[b][s*2+n][h][w0 + 8*hi + 0..7], two aligned float4s per lane
        if (n < On) {
            float* op = out + obase + (long)(s * 2 + n) * hw;
            *(float4*)(op + 0) = make_float4(spk[0], spk[1], spk[2], spk[3]);
            *(float4*)(op + 4) = make_float4(spk[4], spk[5], spk[6], spk[7]);
        }
    }
}

extern "C" void kernel_launch(void* const* d_in, const int* in_sizes, int n_in,
                              void* d_out, int out_size, void* d_ws, size_t ws_size,
                              hipStream_t stream) {
    const float* all_input = (const float*)d_in[0];
    const float* conv_w    = (const float*)d_in[1];
    const float* conv_b    = (const float*)d_in[2];
    const float* gamma     = (const float*)d_in[3];
    const float* beta_bn   = (const float*)d_in[4];
    const float* run_mean  = (const float*)d_in[5];
    const float* run_var   = (const float*)d_in[6];
    const float* rho       = (const float*)d_in[7];
    float* out = (float*)d_out;

    const int n_segments = Bn * Hn * SEGS_PER_ROW;           // 262144
    const int n_blocks   = n_segments / WAVES_PER_BLOCK;     // 32768
    hipLaunchKernelGGL(snn_front_kernel, dim3(n_blocks), dim3(256), 0, stream,
                       all_input, conv_w, conv_b, gamma, beta_bn,
                       run_mean, run_var, rho, out);
}
